// ConvNd_19739669693183
// MI455X (gfx1250) — matrix operand order repaired and checked by reading, not compiled, not run on hardware
//
#include <hip/hip_runtime.h>

// ---------------------------------------------------------------------------
// 3D conv (B=4, Cin=Cout=64, 32^3, 3x3x3, pad 1) as implicit GEMM on CDNA5
// bf16 WMMA with fp32-accurate bf16x3 split.
// Round 3: packed x volume carries a zero halo (34^3, NDHWC) so inner-loop
// A-fragment loads are unconditional b128s -> no exec masking / branches in
// the hot loop; body is pure global_load_b128 + v_wmma_f32_16x16x32_bf16.
// ---------------------------------------------------------------------------

typedef __attribute__((ext_vector_type(16))) __bf16 v16bf;
typedef __attribute__((ext_vector_type(8)))  __bf16 v8bf;   // 16 bytes
typedef __attribute__((ext_vector_type(8)))  float  v8f;

#define CIN   64
#define COUT  64
#define DSP   32
#define HSP   32
#define WSP   32
#define SPVOL (DSP * HSP * WSP)          // 32768
#define NB    4
#define PD    34                          // padded spatial dim (halo of 1)
#define PVOX  (PD * PD * PD)              // 39304 voxels per batch

// ---- workspace layout (bytes) ---------------------------------------------
// xpk: padded NDHWC bf16; element index (((b*PD+zp)*PD+yp)*PD+xp)*CIN + ci
#define XPK_ELEMS ((size_t)NB * PVOX * CIN)           // 10,061,824
// wpk: per-lane B-fragment records: (((tap*2+chunk)*4+t)*32+lane)*16 + e
#define WPK_ELEMS ((size_t)27 * 2 * 4 * 32 * 16)      // 110,592
#define OFF_XHI ((size_t)0)
#define OFF_XLO (OFF_XHI + XPK_ELEMS * 2)
#define OFF_WHI (OFF_XLO + XPK_ELEMS * 2)
#define OFF_WLO (OFF_WHI + WPK_ELEMS * 2)
// total ws: ~40.7 MB

union FragBF {
    v16bf v;
    v8bf  h[2];
};

// ---------------------------------------------------------------------------
// Pre-pass 1: x (NCDHW f32) -> padded NDHWC bf16 hi/lo with zero halo.
// tid order == output element order -> fully coalesced writes.
// ---------------------------------------------------------------------------
__global__ __launch_bounds__(256) void pack_x_pad(
    const float* __restrict__ x, __bf16* __restrict__ xhi,
    __bf16* __restrict__ xlo)
{
    const size_t tid = (size_t)blockIdx.x * 256 + threadIdx.x;
    const int    ci  = (int)(tid & 63);
    size_t vox = tid >> 6;
    const int xp = (int)(vox % PD); vox /= PD;
    const int yp = (int)(vox % PD); vox /= PD;
    const int zp = (int)(vox % PD);
    const int b  = (int)(vox / PD);

    const bool interior = (xp >= 1) && (xp <= WSP) &&
                          (yp >= 1) && (yp <= HSP) &&
                          (zp >= 1) && (zp <= DSP);
    float v = 0.f;
    if (interior) {
        v = x[((size_t)b * CIN + ci) * SPVOL
              + (size_t)(zp - 1) * (HSP * WSP) + (yp - 1) * WSP + (xp - 1)];
    }
    __bf16 h = (__bf16)v;
    __bf16 l = (__bf16)(v - (float)h);
    xhi[tid] = h;
    xlo[tid] = l;
}

// ---------------------------------------------------------------------------
// Pre-pass 2: w (OIDHW f32) -> per-lane B-fragment bf16 records, hi + lo.
// B-matrix 32x16 wave32 layout: element e of lane (n=lane&15, hi=lane>>4)
// holds K = (e>>3)*16 + hi*8 + (e&7), N = n.
// ---------------------------------------------------------------------------
__global__ __launch_bounds__(256) void pack_w(
    const float* __restrict__ w, __bf16* __restrict__ whi,
    __bf16* __restrict__ wlo)
{
    int tid = blockIdx.x * 256 + threadIdx.x;   // exactly 110,592 threads
    int e     =  tid        & 15;
    int lane  = (tid >> 4)  & 31;
    int t     = (tid >> 9)  & 3;
    int chunk = (tid >> 11) & 1;
    int tap   =  tid >> 12;                      // 0..26
    int ci = chunk * 32 + (e >> 3) * 16 + (lane >> 4) * 8 + (e & 7);
    int co = t * 16 + (lane & 15);
    float  v = w[((size_t)co * CIN + ci) * 27 + tap];
    __bf16 h = (__bf16)v;
    __bf16 l = (__bf16)(v - (float)h);
    whi[tid] = h;
    wlo[tid] = l;
}

// ---------------------------------------------------------------------------
// Main kernel: one wave32 per (b,z,y, 16-wide x strip), all 64 cout channels.
// Inner body per (tap, cin-chunk): 4 unconditional b128 A-loads, 16 b128
// B-loads, 12 x v_wmma_f32_16x16x32_bf16 (hi*hi + hi*lo + lo*hi, 4 N-tiles).
// EXEC stays all-1s at every WMMA (only wave-uniform branches).
// ---------------------------------------------------------------------------
__global__ __launch_bounds__(256) void conv3d_wmma_bf16x3(
    const __bf16* __restrict__ xhi, const __bf16* __restrict__ xlo,
    const __bf16* __restrict__ whi, const __bf16* __restrict__ wlo,
    const float*  __restrict__ bias, float* __restrict__ out)
{
    const int lane = threadIdx.x & 31;
    const int wid  = blockIdx.x * 8 + (threadIdx.x >> 5);  // 0..8191, exact

    const int x0 = (wid & 1) << 4;
    const int y  = (wid >> 1) & 31;
    const int z  = (wid >> 6) & 31;
    const int b  =  wid >> 11;

    const int m  = lane & 15;   // A row (spatial); also B/C column index
    const int hi = lane >> 4;   // lane half

    v8f acc[4];
    #pragma unroll
    for (int t = 0; t < 4; ++t) acc[t] = (v8f){0.f,0.f,0.f,0.f,0.f,0.f,0.f,0.f};

    for (int kz = 0; kz < 3; ++kz) {
        const int zi = z + kz - 1;
        if (zi < 0 || zi >= DSP) continue;          // wave-uniform: tap plane all-zero
        for (int ky = 0; ky < 3; ++ky) {
            const int yi = y + ky - 1;
            if (yi < 0 || yi >= HSP) continue;      // wave-uniform skip

            // padded voxel row: zp = z+kz, yp = y+ky (halo offset absorbed)
            const size_t prow =
                (((size_t)b * PD + (z + kz)) * PD + (y + ky)) * PD;

            for (int kx = 0; kx < 3; ++kx) {
                const int tap = kz * 9 + ky * 3 + kx;
                // per-lane padded voxel record (64 bf16, 128B aligned);
                // xp = x0 + m + kx in [0,33] -- always in-bounds, halo = 0
                const size_t vox = (prow + (size_t)(x0 + m + kx)) * CIN;

                #pragma unroll
                for (int chunk = 0; chunk < 2; ++chunk) {
                    const size_t run0 = vox + (size_t)(chunk * 32 + 8 * hi);

                    // ---- A fragments: two contiguous 8-bf16 runs, hi & lo
                    FragBF ah, al;
                    ah.h[0] = *(const v8bf*)(xhi + run0);
                    ah.h[1] = *(const v8bf*)(xhi + run0 + 16);
                    al.h[0] = *(const v8bf*)(xlo + run0);
                    al.h[1] = *(const v8bf*)(xlo + run0 + 16);

                    // ---- 4 N-tiles of cout, reuse A fragments
                    #pragma unroll
                    for (int t = 0; t < 4; ++t) {
                        const size_t rec =
                            ((((size_t)tap * 2 + chunk) * 4 + t) * 32 + lane) * 16;
                        FragBF wh, wl;
                        wh.h[0] = *(const v8bf*)(whi + rec);
                        wh.h[1] = *(const v8bf*)(whi + rec + 8);
                        wl.h[0] = *(const v8bf*)(wlo + rec);
                        wl.h[1] = *(const v8bf*)(wlo + rec + 8);

                        acc[t] = __builtin_amdgcn_wmma_f32_16x16x32_bf16(
                            false, ah.v, false, wh.v, (short)0, acc[t], false, false);
                        acc[t] = __builtin_amdgcn_wmma_f32_16x16x32_bf16(
                            false, ah.v, false, wl.v, (short)0, acc[t], false, false);
                        acc[t] = __builtin_amdgcn_wmma_f32_16x16x32_bf16(
                            false, al.v, false, wh.v, (short)0, acc[t], false, false);
                    }
                }
            }
        }
    }

    // ---- epilogue: C/D layout VGPR r -> M = r + 8*hi, N = lane&15.
    // r is contiguous in output x -> two 16B vector stores per tile.
    const int n = lane & 15;
    #pragma unroll
    for (int t = 0; t < 4; ++t) {
        const int   co = t * 16 + n;
        const float bv = bias[co];
        const size_t obase = ((size_t)b * COUT + co) * SPVOL
                           + (size_t)z * (HSP * WSP) + (size_t)y * WSP
                           + (size_t)(x0 + 8 * hi);
        float4 s0 = make_float4(acc[t][0] + bv, acc[t][1] + bv,
                                acc[t][2] + bv, acc[t][3] + bv);
        float4 s1 = make_float4(acc[t][4] + bv, acc[t][5] + bv,
                                acc[t][6] + bv, acc[t][7] + bv);
        *(float4*)(out + obase)     = s0;
        *(float4*)(out + obase + 4) = s1;
    }
}

extern "C" void kernel_launch(void* const* d_in, const int* in_sizes, int n_in,
                              void* d_out, int out_size, void* d_ws, size_t ws_size,
                              hipStream_t stream) {
    (void)in_sizes; (void)n_in; (void)out_size; (void)ws_size;
    const float* x    = (const float*)d_in[0];
    const float* w    = (const float*)d_in[1];
    const float* bias = (const float*)d_in[2];
    float* out        = (float*)d_out;

    char* ws = (char*)d_ws;
    __bf16* xhi = (__bf16*)(ws + OFF_XHI);
    __bf16* xlo = (__bf16*)(ws + OFF_XLO);
    __bf16* whi = (__bf16*)(ws + OFF_WHI);
    __bf16* wlo = (__bf16*)(ws + OFF_WLO);

    // Pre-pass: halo-padded bf16 split of x (10,061,824 elems, exact grid)
    pack_x_pad<<<dim3((unsigned)(XPK_ELEMS / 256)), dim3(256), 0, stream>>>(x, xhi, xlo);
    // Pre-pass: weight fragment swizzle (110,592 elems, exact grid)
    pack_w<<<dim3((unsigned)(WPK_ELEMS / 256)), dim3(256), 0, stream>>>(w, whi, wlo);

    // Main implicit-GEMM conv: 8192 waves = 1024 blocks x 8 wave32s
    conv3d_wmma_bf16x3<<<dim3(1024), dim3(256), 0, stream>>>(
        xhi, xlo, whi, wlo, bias, out);
}
